// LinearAttention2D_50989851738213
// MI455X (gfx1250) — compile-verified, matrix operand order checked by reading, not run on hardware
//
#include <hip/hip_runtime.h>
#include <cstdint>
#include <cstddef>

// ---------- types ----------
typedef __attribute__((ext_vector_type(16))) __bf16 v16bf;
typedef __attribute__((ext_vector_type(8)))  float  v8f;
typedef __attribute__((ext_vector_type(4)))  int    v4i;

union FragU { unsigned int u32[8]; v16bf v; };

struct __align__(16) U4 { unsigned int x, y, z, w; };

#define GLOBAL_AS __attribute__((address_space(1)))
#define LDS_AS    __attribute__((address_space(3)))

// gfx1250 async global->LDS path (ASYNCcnt-tracked), guarded so absence of the
// builtins falls back to the synchronous VGPR round-trip staging.
#if defined(__AMDGCN__) && __has_builtin(__builtin_amdgcn_global_load_async_to_lds_b128) && __has_builtin(__builtin_amdgcn_s_wait_asynccnt)
#define USE_ASYNC_LDS 1
#else
#define USE_ASYNC_LDS 0
#endif

__device__ __forceinline__ void copy16_to_lds(unsigned short* dst_lds, const unsigned short* src_g) {
#if USE_ASYNC_LDS
    __builtin_amdgcn_global_load_async_to_lds_b128(
        (GLOBAL_AS v4i*)src_g, (LDS_AS v4i*)dst_lds, 0, 0);
#else
    *(U4*)dst_lds = *(const U4*)src_g;
#endif
}

__device__ __forceinline__ void wait_tiles_and_barrier() {
#if USE_ASYNC_LDS
    __builtin_amdgcn_s_wait_asynccnt(0);
#endif
    __syncthreads();
}

__device__ __forceinline__ unsigned short f2bf(float f) {
    unsigned int u = __float_as_uint(f);
    u += 0x7FFFu + ((u >> 16) & 1u);           // round-to-nearest-even
    return (unsigned short)(u >> 16);
}

// k index pattern shared by 16-bit A (16x32) and B (32x16) fragments (ISA 7.12.2):
// lanes 0-15: K in {0..7, 16..23}; lanes 16-31: K in {8..15, 24..31}; packed pairs.
__device__ __forceinline__ int frag_k(int sel, int r) {
    return ((r >> 2) << 4) + sel + ((r & 3) << 1);
}

// ---------- generic bf16 GEMM:  C[M,N] = A[M,K] * B[K,N] (+bias) ----------
// A row-major (lda), B row-major (ldb), C row-major (ldc). BN = 128 fixed.
// Double-buffered LDS; tiles streamed with async global->LDS copies.
template<int BM, bool BIAS, bool OUTF32>
__global__ __launch_bounds__(256) void gemm_bf16_kernel(
    const unsigned short* __restrict__ A,
    const unsigned short* __restrict__ Bm,
    float* __restrict__ Cf,
    unsigned short* __restrict__ Cb,
    const float* __restrict__ bias,
    int Ktot, int lda, int ldb, int ldc,
    long long aStride, long long bStride, long long cStride)
{
    constexpr int BN    = 128;
    constexpr int WROWS = (BM == 128) ? 2 : 1;     // waves along M (wave M-extent 64)
    constexpr int WN    = (BN * WROWS) / 8;        // 32 or 16
    constexpr int NSUB  = WN / 16;                 // 2 or 1

    __shared__ unsigned short As[2][BM * 32];
    __shared__ unsigned short Bs[2][32 * BN];

    const int tid  = threadIdx.x;
    const int lane = tid & 31;
    const int wave = tid >> 5;
    const int wm   = (WROWS == 2) ? (wave & 1) : 0;
    const int wn   = (WROWS == 2) ? (wave >> 1) : wave;
    const int m0   = wm * 64;
    const int n0   = wn * WN;

    const int m_blk = blockIdx.y * BM;
    const int n_blk = blockIdx.x * BN;

    const unsigned short* Ab = A  + (long long)blockIdx.z * aStride;
    const unsigned short* Bb = Bm + (long long)blockIdx.z * bStride;

    const v8f vzero = {0.f, 0.f, 0.f, 0.f, 0.f, 0.f, 0.f, 0.f};
    v8f acc[4][NSUB];
    #pragma unroll
    for (int i = 0; i < 4; ++i)
        #pragma unroll
        for (int j = 0; j < NSUB; ++j) acc[i][j] = vzero;

    const int lm  = lane & 15;
    const int sel = (lane >> 4) << 3;   // 0 or 8

    constexpr int LA = (BM * 32) / (256 * 8);      // 16B chunks per thread for A tile
    auto stage = [&](int kk, int bb) {
        #pragma unroll
        for (int i = 0; i < LA; ++i) {
            const int l = (tid + i * 256) * 8;
            const int r = l >> 5, c = l & 31;
            copy16_to_lds(&As[bb][l], &Ab[(long long)(m_blk + r) * lda + kk + c]);
        }
        #pragma unroll
        for (int i = 0; i < 2; ++i) {              // 32*128 / (256*8) = 2
            const int l = (tid + i * 256) * 8;
            const int r = l >> 7, c = l & 127;
            copy16_to_lds(&Bs[bb][l], &Bb[(long long)(kk + r) * ldb + n_blk + c]);
        }
    };

    const int nsteps = Ktot >> 5;
    stage(0, 0);

    for (int s = 0; s < nsteps; ++s) {
        wait_tiles_and_barrier();                  // tile s now visible to all waves
        if (s + 1 < nsteps) stage((s + 1) * 32, (s + 1) & 1);
        if (s + 2 < nsteps)                        // gfx1250 global_prefetch_b8
            __builtin_prefetch(&Bb[(long long)((s + 2) * 32 + (tid >> 3)) * ldb + n_blk + (tid & 7) * 16], 0, 0);

        const unsigned short* Ac = As[s & 1];
        const unsigned short* Bc = Bs[s & 1];

        // ---- build B fragments once per K-step ----
        FragU bf_[NSUB];
        #pragma unroll
        for (int j = 0; j < NSUB; ++j) {
            const int col = n0 + j * 16 + lm;
            #pragma unroll
            for (int r = 0; r < 8; ++r) {
                const int k = frag_k(sel, r);
                unsigned int lo = Bc[k * BN + col];
                unsigned int hi = Bc[(k + 1) * BN + col];
                bf_[j].u32[r] = lo | (hi << 16);
            }
        }
        // ---- A fragments + WMMA ----
        #pragma unroll
        for (int i = 0; i < 4; ++i) {
            FragU a;
            const int row = m0 + i * 16 + lm;
            #pragma unroll
            for (int r = 0; r < 8; ++r)
                a.u32[r] = *(const unsigned int*)&Ac[row * 32 + frag_k(sel, r)];
            #pragma unroll
            for (int j = 0; j < NSUB; ++j)
                acc[i][j] = __builtin_amdgcn_wmma_f32_16x16x32_bf16(
                    false, a.v, false, bf_[j].v, (short)0, acc[i][j], false, false);
        }
    }

    // ---- store (C/D layout: VGPR t -> M = t + 8*(lane>=16), N = lane&15) ----
    const int rowoff = (lane >> 4) << 3;
    const long long cbase = (long long)blockIdx.z * cStride;
    #pragma unroll
    for (int i = 0; i < 4; ++i)
        #pragma unroll
        for (int j = 0; j < NSUB; ++j)
            #pragma unroll
            for (int t = 0; t < 8; ++t) {
                const int gm = m_blk + m0 + i * 16 + rowoff + t;
                const int gn = n_blk + n0 + j * 16 + lm;
                float val = acc[i][j][t];
                if (BIAS) val += bias[gm];
                if (OUTF32) Cf[cbase + (long long)gm * ldc + gn] = val;
                else        Cb[cbase + (long long)gm * ldc + gn] = f2bf(val);
            }
}

// ---------- stage 3: context[d,e] = sum_n k[d,n]*v[e,n]; store ctx^T (e x d) bf16 ----------
__global__ __launch_bounds__(256) void ctx_kernel(
    const unsigned short* __restrict__ kbf,
    const unsigned short* __restrict__ vbf,
    unsigned short* __restrict__ ctx)
{
    __shared__ unsigned short Kt[2][64 * 32];
    __shared__ unsigned short Vt[2][64 * 32];
    const int bh = blockIdx.x;
    const unsigned short* Kp = kbf + (long long)bh * 64 * 4096;
    const unsigned short* Vp = vbf + (long long)bh * 64 * 4096;
    const int tid = threadIdx.x, lane = tid & 31, wave = tid >> 5;
    const int dsub = wave & 3, ep = wave >> 2;
    const int lm = lane & 15, sel = (lane >> 4) << 3;

    const v8f vzero = {0.f, 0.f, 0.f, 0.f, 0.f, 0.f, 0.f, 0.f};
    v8f acc[2] = {vzero, vzero};

    auto stage = [&](int nn, int bb) {
        const int l = tid * 8, r = l >> 5, c = l & 31;
        copy16_to_lds(&Kt[bb][l], &Kp[(long long)r * 4096 + nn + c]);
        copy16_to_lds(&Vt[bb][l], &Vp[(long long)r * 4096 + nn + c]);
    };

    stage(0, 0);
    for (int s = 0; s < 128; ++s) {
        wait_tiles_and_barrier();
        if (s + 1 < 128) stage((s + 1) * 32, (s + 1) & 1);

        const unsigned short* Kc = Kt[s & 1];
        const unsigned short* Vc = Vt[s & 1];

        FragU a;
        const int d = dsub * 16 + lm;
        #pragma unroll
        for (int r2 = 0; r2 < 8; ++r2)
            a.u32[r2] = *(const unsigned int*)&Kc[d * 32 + frag_k(sel, r2)];
        #pragma unroll
        for (int j = 0; j < 2; ++j) {
            FragU b;
            const int e = ep * 32 + j * 16 + lm;
            #pragma unroll
            for (int r2 = 0; r2 < 8; ++r2)   // B[k=n][col=e] = v[e][n] -> contiguous pairs in Vt row
                b.u32[r2] = *(const unsigned int*)&Vc[e * 32 + frag_k(sel, r2)];
            acc[j] = __builtin_amdgcn_wmma_f32_16x16x32_bf16(
                false, a.v, false, b.v, (short)0, acc[j], false, false);
        }
    }
    const int rowoff = (lane >> 4) << 3;
    #pragma unroll
    for (int j = 0; j < 2; ++j)
        #pragma unroll
        for (int t = 0; t < 8; ++t) {
            const int d = dsub * 16 + rowoff + t;
            const int e = ep * 32 + j * 16 + lm;
            ctx[(long long)bh * 4096 + e * 64 + d] = f2bf(acc[j][t]);
        }
}

// ---------- stage 2 ----------
__global__ __launch_bounds__(256) void qsoftmax_kernel(const float* __restrict__ qkv,
                                                       unsigned short* __restrict__ qbf)
{
    const int idx = blockIdx.x * 256 + threadIdx.x;   // b*16384 + h*4096 + n
    const int n = idx & 4095;
    const int h = (idx >> 12) & 3;
    const int b = idx >> 14;
    const float* src = qkv + ((long long)b * 768 + h * 64) * 4096 + n;
    unsigned short* dst = qbf + ((long long)b * 256 + h * 64) * 4096 + n;
    float v[64]; float mx = -1e30f;
    #pragma unroll
    for (int d = 0; d < 64; ++d) { v[d] = src[(long long)d * 4096]; mx = fmaxf(mx, v[d]); }
    float sum = 0.f;
    #pragma unroll
    for (int d = 0; d < 64; ++d) { v[d] = __expf(v[d] - mx); sum += v[d]; }
    const float s = 0.125f / sum;                     // SCALE = 64^-0.5
    #pragma unroll
    for (int d = 0; d < 64; ++d) dst[(long long)d * 4096] = f2bf(v[d] * s);
}

__global__ __launch_bounds__(256) void ksoftmax_kernel(const float* __restrict__ qkv,
                                                       unsigned short* __restrict__ kbf)
{
    __shared__ float red[256];
    const int blk = blockIdx.x;                       // b*256 + r
    const int b = blk >> 8, r = blk & 255;
    const float* src = qkv + ((long long)b * 768 + 256 + r) * 4096;
    unsigned short* dst = kbf + ((long long)b * 256 + r) * 4096;
    const int tid = threadIdx.x;
    float v[16]; float mx = -1e30f;
    #pragma unroll
    for (int j = 0; j < 16; ++j) { v[j] = src[tid + j * 256]; mx = fmaxf(mx, v[j]); }
    red[tid] = mx; __syncthreads();
    for (int s = 128; s > 0; s >>= 1) { if (tid < s) red[tid] = fmaxf(red[tid], red[tid + s]); __syncthreads(); }
    mx = red[0]; __syncthreads();
    float sum = 0.f;
    #pragma unroll
    for (int j = 0; j < 16; ++j) { v[j] = __expf(v[j] - mx); sum += v[j]; }
    red[tid] = sum; __syncthreads();
    for (int s = 128; s > 0; s >>= 1) { if (tid < s) red[tid] += red[tid + s]; __syncthreads(); }
    const float inv = 1.f / red[0];
    #pragma unroll
    for (int j = 0; j < 16; ++j) dst[tid + j * 256] = f2bf(v[j] * inv);
}

__global__ void vconvert_kernel(const float* __restrict__ qkv, unsigned short* __restrict__ vbf)
{
    const long long idx = (long long)blockIdx.x * 256 + threadIdx.x;  // B*256*4096 = 2^24
    const long long b = idx >> 20, rem = idx & 0xFFFFFLL;
    vbf[idx] = f2bf(qkv[b * (768LL * 4096) + 512LL * 4096 + rem]);
}

__global__ void f2bf_kernel(const float* __restrict__ src, unsigned short* __restrict__ dst, long long n)
{
    const long long i = (long long)blockIdx.x * 256 + threadIdx.x;
    if (i < n) dst[i] = f2bf(src[i]);
}

// ---------- stage 6: GroupNorm ----------
__global__ __launch_bounds__(256) void gn_reduce_kernel(const float* __restrict__ z, float* __restrict__ stats)
{
    __shared__ float rs[256], rq[256];
    const int bg = blockIdx.x, tid = threadIdx.x;
    const float* base = z + (long long)bg * 262144;   // 64 ch * 4096, contiguous per group
    float s = 0.f, q = 0.f;
    for (int i = tid; i < 262144; i += 256) { const float x = base[i]; s += x; q += x * x; }
    rs[tid] = s; rq[tid] = q; __syncthreads();
    for (int st = 128; st > 0; st >>= 1) {
        if (tid < st) { rs[tid] += rs[tid + st]; rq[tid] += rq[tid + st]; }
        __syncthreads();
    }
    if (tid == 0) {
        const float mean = rs[0] * (1.f / 262144.f);
        const float var  = rq[0] * (1.f / 262144.f) - mean * mean;
        stats[bg * 2 + 0] = mean;
        stats[bg * 2 + 1] = rsqrtf(var + 1e-5f);
    }
}

__global__ void gn_apply_kernel(const float* __restrict__ z, const float* __restrict__ stats,
                                const float* __restrict__ gw, const float* __restrict__ gb,
                                float* __restrict__ out)
{
    const long long idx = (long long)blockIdx.x * 256 + threadIdx.x;  // 2^24
    const int c  = (int)((idx >> 12) & 255);
    const int bg = (int)(idx >> 18);
    const float mean = stats[bg * 2], inv = stats[bg * 2 + 1];
    out[idx] = (z[idx] - mean) * inv * gw[c] + gb[c];
}

// ---------- host ----------
extern "C" void kernel_launch(void* const* d_in, const int* in_sizes, int n_in,
                              void* d_out, int out_size, void* d_ws, size_t ws_size,
                              hipStream_t stream)
{
    (void)in_sizes; (void)n_in; (void)out_size; (void)ws_size;
    const float* x     = (const float*)d_in[0];
    const float* w_qkv = (const float*)d_in[1];
    const float* w_out = (const float*)d_in[2];
    const float* b_out = (const float*)d_in[3];
    const float* gn_w  = (const float*)d_in[4];
    const float* gn_b  = (const float*)d_in[5];
    float* out = (float*)d_out;

    char* ws = (char*)d_ws;
    size_t off = 0;
    auto alloc = [&](size_t bytes) -> void* {
        void* p = (void*)(ws + off);
        off += (bytes + 255) & ~(size_t)255;
        return p;
    };

    unsigned short* x_bf    = (unsigned short*)alloc(16ULL * 256 * 4096 * 2);
    unsigned short* wqkv_bf = (unsigned short*)alloc(768ULL * 256 * 2);
    unsigned short* wout_bf = (unsigned short*)alloc(256ULL * 256 * 2);
    float*          qkv     = (float*)         alloc(16ULL * 768 * 4096 * 4);
    unsigned short* q_bf    = (unsigned short*)alloc(16ULL * 256 * 4096 * 2);
    unsigned short* k_bf    = (unsigned short*)alloc(16ULL * 256 * 4096 * 2);
    unsigned short* v_bf    = (unsigned short*)alloc(16ULL * 256 * 4096 * 2);
    unsigned short* ctx_bf  = (unsigned short*)alloc(64ULL * 64 * 64 * 2);
    unsigned short* ao_bf   = (unsigned short*)alloc(16ULL * 256 * 4096 * 2);
    float*          z       = (float*)         alloc(16ULL * 256 * 4096 * 4);
    float*          stats   = (float*)         alloc(64ULL * 2 * 4);

    // inputs -> bf16
    f2bf_kernel<<<65536, 256, 0, stream>>>(x, x_bf, 16LL * 256 * 4096);
    f2bf_kernel<<<768, 256, 0, stream>>>(w_qkv, wqkv_bf, 768LL * 256);
    f2bf_kernel<<<256, 256, 0, stream>>>(w_out, wout_bf, 256LL * 256);

    // stage 1: qkv = w_qkv @ x   (M=768, N=4096, K=256 per batch)
    gemm_bf16_kernel<128, false, true><<<dim3(32, 6, 16), 256, 0, stream>>>(
        wqkv_bf, x_bf, qkv, nullptr, nullptr,
        256, 256, 4096, 4096, 0LL, 256LL * 4096, 768LL * 4096);

    // stage 2: softmaxes + v pass-through
    qsoftmax_kernel<<<1024, 256, 0, stream>>>(qkv, q_bf);
    ksoftmax_kernel<<<16 * 256, 256, 0, stream>>>(qkv, k_bf);
    vconvert_kernel<<<65536, 256, 0, stream>>>(qkv, v_bf);

    // stage 3: context per (b,h)
    ctx_kernel<<<64, 256, 0, stream>>>(k_bf, v_bf, ctx_bf);

    // stage 4: attn_out = ctx^T @ q   (M=64, N=4096, K=64 per (b,h))
    gemm_bf16_kernel<64, false, false><<<dim3(32, 1, 64), 256, 0, stream>>>(
        ctx_bf, q_bf, nullptr, ao_bf, nullptr,
        64, 64, 4096, 4096, 4096LL, 64LL * 4096, 64LL * 4096);

    // stage 5: z = w_out @ attn_out + b_out   (M=256, N=4096, K=256 per batch)
    gemm_bf16_kernel<128, true, true><<<dim3(32, 2, 16), 256, 0, stream>>>(
        wout_bf, ao_bf, z, nullptr, b_out,
        256, 256, 4096, 4096, 0LL, 256LL * 4096, 256LL * 4096);

    // stage 6: GroupNorm
    gn_reduce_kernel<<<64, 256, 0, stream>>>(z, stats);
    gn_apply_kernel<<<65536, 256, 0, stream>>>(z, stats, gn_w, gn_b, out);
}
